// MultiHeadAttention_68865505624655
// MI455X (gfx1250) — matrix operand order repaired and checked by reading, not compiled, not run on hardware
//
#include <hip/hip_runtime.h>
#include <hip/hip_bf16.h>

// ---------------------------------------------------------------------------
// Problem constants (B=C=1)
// ---------------------------------------------------------------------------
#define T_SEQ   3072
#define D_MODEL 1024
#define H_HEADS 16
#define X_HEAD  64
#define THW     49152   // T_SEQ * H_HEADS

typedef __bf16 bf16_t;
typedef __attribute__((ext_vector_type(16))) __bf16 v16bf;
typedef __attribute__((ext_vector_type(8)))  __bf16 v8bf;
typedef __attribute__((ext_vector_type(8)))  float  v8f;

// float -> bf16 with round-to-nearest-even (bit ops: no reliance on __bf16 casts)
__device__ __forceinline__ bf16_t f2bf(float f) {
    unsigned u = __builtin_bit_cast(unsigned, f);
    unsigned r = (u + 0x7fffu + ((u >> 16) & 1u)) >> 16;
    unsigned short s = (unsigned short)r;
    return __builtin_bit_cast(bf16_t, s);
}

// assemble a 16-element bf16 fragment from two 16-byte chunks
__device__ __forceinline__ v16bf load_frag16(const bf16_t* p0, const bf16_t* p1) {
    v8bf a = *(const v8bf*)p0;
    v8bf b = *(const v8bf*)p1;
    v16bf r;
#pragma unroll
    for (int i = 0; i < 8; ++i) { r[i] = a[i]; r[i + 8] = b[i]; }
    return r;
}

__device__ __forceinline__ v8f wmma_bf16(v16bf a, v16bf b, v8f c) {
    // D = A(16x32 bf16) x B(32x16 bf16) + C(16x16 f32)
    return __builtin_amdgcn_wmma_f32_16x16x32_bf16(
        /*neg_a=*/false, a, /*neg_b=*/false, b,
        /*c_mod=*/(short)0, c, /*reuse_a=*/false, /*reuse_b=*/false);
}

// CDNA5 async copy: 16 bytes per lane, global -> LDS, tracked by ASYNCcnt.
// Generic-pointer low 32 bits == LDS byte offset (ISA 10.2 aperture rule).
__device__ __forceinline__ void async_copy_b128(void* lds, const void* g) {
    unsigned l = (unsigned)(size_t)lds;
    asm volatile("global_load_async_to_lds_b128 %0, %1, off"
                 :: "v"(l), "v"(g) : "memory");
}
__device__ __forceinline__ void wait_asynccnt0() {
    asm volatile("s_wait_asynccnt 0x0" ::: "memory");
}

// reductions across the 16 lanes sharing a C-matrix row group (xor masks 1..8
// stay within lanes [0,15] / [16,31], matching the wave32 WMMA C layout)
__device__ __forceinline__ float redmax16(float v) {
#pragma unroll
    for (int m = 1; m <= 8; m <<= 1) v = fmaxf(v, __shfl_xor(v, m, 32));
    return v;
}
__device__ __forceinline__ float redsum16(float v) {
#pragma unroll
    for (int m = 1; m <= 8; m <<= 1) v += __shfl_xor(v, m, 32);
    return v;
}

// ---------------------------------------------------------------------------
// Kernel 1: fp32 -> bf16 conversion for x and the three weight matrices
// grid: 12288 x 256  (exactly 3145728 threads; |x| == |Wq|+|Wk|+|Wv|)
// ---------------------------------------------------------------------------
__global__ __launch_bounds__(256) void convert_kernel(
    const float* __restrict__ x,
    const float* __restrict__ Wq, const float* __restrict__ Wk,
    const float* __restrict__ Wv,
    bf16_t* __restrict__ xb, bf16_t* __restrict__ Wb)
{
    int i = blockIdx.x * 256 + threadIdx.x;     // 0 .. 3145727
    xb[i] = f2bf(x[i]);
    int z = i >> 20;                            // which weight matrix
    int r = i & 0xFFFFF;                        // element within it
    const float* W = (z == 0) ? Wq : (z == 1) ? Wk : Wv;
    Wb[i] = f2bf(W[r]);
}

// ---------------------------------------------------------------------------
// Kernel 2: projection GEMM  P = x @ W^T + b  (bf16 WMMA, f32 accum)
// then scatter into de-scrambled per-head layouts:
//   Qh[h][t2][x], Kh[h][t2][x]  (row-major, 64-wide rows)
//   Vt[h][x][t2]                (transposed: contiguous over kv)
// grid: (1536, 1, 3) x 256 ; wave w of block b computes output tile b*8+w
// ---------------------------------------------------------------------------
__global__ __launch_bounds__(256) void proj_kernel(
    const bf16_t* __restrict__ xb, const bf16_t* __restrict__ Wb,
    const float* __restrict__ bq, const float* __restrict__ bk,
    const float* __restrict__ bv,
    bf16_t* __restrict__ Qh, bf16_t* __restrict__ Kh, bf16_t* __restrict__ Vg)
{
    const int wave = threadIdx.x >> 5;
    const int lane = threadIdx.x & 31;
    const int half = lane >> 4;
    const int n0   = lane & 15;
    const int koff = half * 8;

    const int tile = blockIdx.x * 8 + wave;     // 0 .. 12287
    const int t0 = (tile >> 6) * 16;            // 192 t-tiles
    const int e0 = (tile & 63) * 16;            // 64  e-tiles
    const int z  = blockIdx.z;                  // 0=Q 1=K 2=V

    const bf16_t* W    = Wb + (size_t)z * 1048576;
    const bf16_t* arow = xb + (size_t)(t0 + n0) * D_MODEL;  // A: row m=lane&15
    const bf16_t* brow = W  + (size_t)(e0 + n0) * D_MODEL;  // B: col n=lane&15 -> W row

    v8f c = {};
    for (int k = 0; k < D_MODEL; k += 32) {
        // A fragment: 16 contiguous K values at [m][k + half*16]
        v16bf A = load_frag16(arow + k + half * 16, arow + k + half * 16 + 8);
        // B fragment: 16 contiguous K values at [n][k + half*16]
        v16bf B = load_frag16(brow + k + half * 16, brow + k + half * 16 + 8);
        c = wmma_bf16(A, B, c);
    }

    const float* bias = (z == 0) ? bq : (z == 1) ? bk : bv;
    const float bval = bias[e0 + n0];

#pragma unroll
    for (int j = 0; j < 8; ++j) {
        int t = t0 + j + 8 * half;              // C layout: M = j + 8*(lane>=16)
        int e = e0 + n0;                        //           N = lane&15
        float val = c[j] + bval;
        // de-scramble: flat = t*D + e = t2*16 + h + x*49152
        unsigned flat = (unsigned)t * 1024u + (unsigned)e;
        unsigned xid  = flat / 49152u;
        unsigned rem  = flat - xid * 49152u;
        unsigned t2   = rem >> 4;
        unsigned hh   = rem & 15u;
        bf16_t b = f2bf(val);
        if (z == 0)      Qh[(((size_t)hh * T_SEQ + t2) << 6) + xid] = b;
        else if (z == 1) Kh[(((size_t)hh * T_SEQ + t2) << 6) + xid] = b;
        else             Vg[(((size_t)hh << 6) + xid) * T_SEQ + t2] = b;
    }
}

// ---------------------------------------------------------------------------
// Kernel 3: fused flash attention per head + gamma*out + x residual.
// Block = 8 waves x 16 query rows. K/V tiles (32 kv steps) are staged into
// LDS once per block via global_load_async_to_lds_b128, double-buffered so
// the async DMA of tile i+1 overlaps compute of tile i.
// grid: (T/128, H) x 256
// ---------------------------------------------------------------------------
__global__ __launch_bounds__(256) void attn_kernel(
    const bf16_t* __restrict__ Qh, const bf16_t* __restrict__ Kh,
    const bf16_t* __restrict__ Vg,
    const float* __restrict__ xin, const float* __restrict__ gamma_p,
    float* __restrict__ y)
{
    __shared__ __align__(16) bf16_t Ktile[2][32][64];  // 2 x 4 KB
    __shared__ __align__(16) bf16_t Vtile[2][64][32];  // 2 x 4 KB
    __shared__ __align__(16) bf16_t Plds[8][16][32];   // per-wave P scratch, 8 KB

    const int tid  = threadIdx.x;
    const int wave = tid >> 5;
    const int lane = tid & 31;
    const int half = lane >> 4;
    const int n0   = lane & 15;
    const int koff = half * 8;
    const int h    = blockIdx.y;
    const int q0   = blockIdx.x * 128 + wave * 16;     // first query row of wave

    const bf16_t* Kbase = Kh + (size_t)h * T_SEQ * X_HEAD;
    const bf16_t* Vbase = Vg + (size_t)h * X_HEAD * T_SEQ;

    // staging-role addresses for this thread (16 B chunk each for K and V)
    const int krow = tid >> 3, kc = (tid & 7) * 8;     // K: 32 rows x 8 chunks
    const int vrow = tid >> 2, vc = (tid & 3) * 8;     // V: 64 rows x 4 chunks

    // persistent Q A-fragments (16 rows x 64 head dims = two K=32 fragments)
    const bf16_t* qrow = Qh + ((size_t)h * T_SEQ + q0 + n0) * X_HEAD;
    v16bf Qa0 = load_frag16(qrow + half * 16,      qrow + half * 16 + 8);
    v16bf Qa1 = load_frag16(qrow + 32 + half * 16, qrow + 32 + half * 16 + 8);

    v8f Oa[4] = {{}, {}, {}, {}};                      // 16x64 f32 accumulator
    float mrow[8], lrow[8];
#pragma unroll
    for (int j = 0; j < 8; ++j) { mrow[j] = -1e30f; lrow[j] = 0.0f; }

    // prologue: stage tile 0 into buffer 0
    async_copy_b128(&Ktile[0][krow][kc], Kbase + (size_t)krow * X_HEAD + kc);
    async_copy_b128(&Vtile[0][vrow][vc], Vbase + (size_t)vrow * T_SEQ + vc);

    const int NT = T_SEQ / 32;                         // 96 kv tiles
    for (int it = 0; it < NT; ++it) {
        const int buf = it & 1;
        wait_asynccnt0();            // this wave's staged chunks are in LDS
        __syncthreads();             // every wave's chunks are in LDS

        if (it + 1 < NT) {           // stage next tile into the other buffer
            const int kvn = (it + 1) * 32;
            async_copy_b128(&Ktile[buf ^ 1][krow][kc],
                            Kbase + (size_t)(kvn + krow) * X_HEAD + kc);
            async_copy_b128(&Vtile[buf ^ 1][vrow][vc],
                            Vbase + (size_t)vrow * T_SEQ + kvn + vc);
        }

        // ---- S = Q K^T for 32 kv columns (two 16-col N tiles), from LDS ---
        const bf16_t* kr0 = &Ktile[buf][n0][0];
        const bf16_t* kr1 = &Ktile[buf][16 + n0][0];
        v16bf Kb00 = load_frag16(kr0 + half * 16,      kr0 + half * 16 + 8);
        v16bf Kb01 = load_frag16(kr0 + 32 + half * 16, kr0 + 32 + half * 16 + 8);
        v16bf Kb10 = load_frag16(kr1 + half * 16,      kr1 + half * 16 + 8);
        v16bf Kb11 = load_frag16(kr1 + 32 + half * 16, kr1 + 32 + half * 16 + 8);

        v8f S0 = {}, S1 = {};
        S0 = wmma_bf16(Qa0, Kb00, S0);
        S0 = wmma_bf16(Qa1, Kb01, S0);
        S1 = wmma_bf16(Qa0, Kb10, S1);
        S1 = wmma_bf16(Qa1, Kb11, S1);

        // ---- online softmax (per row, across the 16-lane half) ------------
        float p0[8], p1[8];
#pragma unroll
        for (int j = 0; j < 8; ++j) {
            float tm = redmax16(fmaxf(S0[j], S1[j]));
            float mn = fmaxf(mrow[j], tm);
            float sc = __expf(mrow[j] - mn);
            mrow[j] = mn;
            p0[j] = __expf(S0[j] - mn);
            p1[j] = __expf(S1[j] - mn);
            float ts = redsum16(p0[j] + p1[j]);
            lrow[j] = lrow[j] * sc + ts;
            Oa[0][j] *= sc; Oa[1][j] *= sc; Oa[2][j] *= sc; Oa[3][j] *= sc;
        }

        // ---- P: C-layout -> A-layout via per-wave LDS round trip ----------
#pragma unroll
        for (int j = 0; j < 8; ++j) {
            Plds[wave][j + 8 * half][n0]      = f2bf(p0[j]);
            Plds[wave][j + 8 * half][n0 + 16] = f2bf(p1[j]);
        }
        const bf16_t* pr = &Plds[wave][n0][0];
        v16bf Pa = load_frag16(pr + half * 16, pr + half * 16 + 8);

        // ---- O += P V  (four 16-col N tiles of the head dim), from LDS ----
#pragma unroll
        for (int n = 0; n < 4; ++n) {
            const bf16_t* vr = &Vtile[buf][n * 16 + n0][half * 16];
            v16bf Vb = load_frag16(vr, vr + 8);
            Oa[n] = wmma_bf16(Pa, Vb, Oa[n]);
        }
    }

    // ---- epilogue: O/l, re-scramble output index, residual ----------------
    const float g = gamma_p[0];
#pragma unroll
    for (int j = 0; j < 8; ++j) {
        const int t2 = q0 + j + 8 * half;
        const float inv = 1.0f / lrow[j];
        const size_t rb = (size_t)t2 * D_MODEL;
#pragma unroll
        for (int n = 0; n < 4; ++n) {
            // y[t2, x*16 + h],  x = n*16 + n0
            const size_t idx = rb + (size_t)(n * 256 + n0 * 16 + h);
            y[idx] = g * (Oa[n][j] * inv) + xin[idx];
        }
    }
}

// ---------------------------------------------------------------------------
// Host-side launcher
// ---------------------------------------------------------------------------
extern "C" void kernel_launch(void* const* d_in, const int* in_sizes, int n_in,
                              void* d_out, int out_size, void* d_ws, size_t ws_size,
                              hipStream_t stream) {
    const float* x     = (const float*)d_in[0];
    const float* Wq    = (const float*)d_in[1];
    const float* bq    = (const float*)d_in[2];
    const float* Wk    = (const float*)d_in[3];
    const float* bk    = (const float*)d_in[4];
    const float* Wv    = (const float*)d_in[5];
    const float* bv    = (const float*)d_in[6];
    const float* gamma = (const float*)d_in[7];
    float* y = (float*)d_out;

    // workspace layout (each region = 3145728 bf16 = 6 MB; total 30 MB)
    char* ws = (char*)d_ws;
    const size_t SEG = 6291456;
    bf16_t* xb = (bf16_t*)(ws + 0 * SEG);
    bf16_t* Wb = (bf16_t*)(ws + 1 * SEG);
    bf16_t* Qh = (bf16_t*)(ws + 2 * SEG);
    bf16_t* Kh = (bf16_t*)(ws + 3 * SEG);
    bf16_t* Vg = (bf16_t*)(ws + 4 * SEG);

    convert_kernel<<<12288, 256, 0, stream>>>(x, Wq, Wk, Wv, xb, Wb);

    dim3 gproj(1536, 1, 3);
    proj_kernel<<<gproj, 256, 0, stream>>>(xb, Wb, bq, bk, bv, Qh, Kh, Vg);

    dim3 gattn(T_SEQ / 128, H_HEADS, 1);
    attn_kernel<<<gattn, 256, 0, stream>>>(Qh, Kh, Vg, x, gamma, y);
}
